// TWC_19473381720808
// MI455X (gfx1250) — matrix operand order repaired and checked by reading, not compiled, not run on hardware
//
#include <hip/hip_runtime.h>
#include <hip/hip_bf16.h>
#include <math.h>

// ---------------------------------------------------------------------------
// FiuriNet recurrent scan on MI455X (gfx1250).
//   B=2048 T=512 N_IN=4 N_HID=64 N_OUT=2 N_GJ=16
// One wave32 owns 16 batch rows and runs the whole T=512 recurrence on the
// matrix pipes, all weights register-resident:
//   - input term (K=4):   v_wmma_f32_16x16x4_f32   (exact f32, 2-VGPR A/B)
//   - recurrent term:     v_wmma_f32_16x16x32_bf16 (K=64 as 2 chained tiles)
//   - gap junctions folded: gather matrix into B_in, leak into decay dg[h]
//   - D-layout -> A-layout hid_O transpose via 2KB LDS tile per step
//   - x_seq software-pipelined: 8 timesteps (4x global_load_b128/lane)
//     prefetched one chunk ahead so no exposed load latency in the chain
// ---------------------------------------------------------------------------

#define BB   2048
#define TT   512
#define HH   64
#define NOUT 2
#define NE   16

typedef __attribute__((ext_vector_type(16))) __bf16 v16bf;
typedef __attribute__((ext_vector_type(8)))  __bf16 v8bf;
typedef __attribute__((ext_vector_type(8)))  float  v8f;
typedef __attribute__((ext_vector_type(2)))  float  v2f;

__device__ __forceinline__ float softplus_f(float x) {
    return x > 20.0f ? x : log1pf(expf(x));
}

// workspace layout:
//   bhid : 8*512 bf16   (tile = kt*4+nt; W_he-W_hi in bf16 B-tile layout)
//   bout : 2*512 bf16   (64x2 padded to 32x16 K-tiles)
//   binf : 4*64  f32    (4x16 f32 B tiles: -W_ii + gj gather matrix)
//   dg   : 64    f32    (sigmoid(dec_hid) - gj leak)

// --------------------------- prologue: weight packing -----------------------
__global__ __launch_bounds__(256) void fiuri_pack(
    const float* __restrict__ w_ii, const float* __restrict__ m_ii,
    const float* __restrict__ w_he, const float* __restrict__ m_he,
    const float* __restrict__ w_hi, const float* __restrict__ m_hi,
    const float* __restrict__ w_ho, const float* __restrict__ m_ho,
    const float* __restrict__ gj_w, const float* __restrict__ dec_hid,
    const int*   __restrict__ gj_src, const int* __restrict__ gj_dst,
    __bf16* __restrict__ bhid, __bf16* __restrict__ bout,
    float* __restrict__ binf, float* __restrict__ dg)
{
    const int tid = threadIdx.x;

    // bf16 B layout (32x16 K-tile): lane L, elem e -> k_local = 16*(L/16)+e,
    // n_local = L%16.
    for (int i = tid; i < 8 * 512; i += 256) {           // bhid: W_he - W_hi
        int tile = i >> 9, lane = (i >> 4) & 31, e = i & 15;
        int kt = tile >> 2, nt = tile & 3;
        int k = kt * 32 + (lane >> 4) * 16 + e;
        int n = nt * 16 + (lane & 15);
        float whe = softplus_f(w_he[k * HH + n]) * m_he[k * HH + n];
        float whi = softplus_f(w_hi[k * HH + n]) * m_hi[k * HH + n];
        bhid[i] = (__bf16)(whe - whi);
    }
    for (int i = tid; i < 2 * 512; i += 256) {           // bout (N padded to 16)
        int tile = i >> 9, lane = (i >> 4) & 31, e = i & 15;
        int k = tile * 32 + (lane >> 4) * 16 + e;
        int n = lane & 15;
        float v = (n < NOUT)
                ? softplus_f(w_ho[k * NOUT + n]) * m_ho[k * NOUT + n] : 0.0f;
        bout[i] = (__bf16)v;
    }
    // f32 B layout (4x16 tile for wmma_f32_16x16x4_f32): lane L, elem j ->
    // k = 2*(L/16)+j, n = L%16.
    for (int i = tid; i < 4 * 64; i += 256) {            // binf: -W_ii + gj
        int tile = i >> 6, lane = (i >> 1) & 31, j = i & 1;
        int k = 2 * (lane >> 4) + j;                     // 0..3, all live
        int n = tile * 16 + (lane & 15);
        float v = -softplus_f(w_ii[k * HH + n]) * m_ii[k * HH + n];
        for (int eg = 0; eg < NE; ++eg)
            if (gj_src[eg] == k && gj_dst[eg] == n)
                v += softplus_f(gj_w[eg]);
        binf[i] = v;
    }
    for (int h = tid; h < HH; h += 256) {                // decay minus gj leak
        float gs = 0.0f;
        for (int eg = 0; eg < NE; ++eg)
            if (gj_dst[eg] == h) gs += softplus_f(gj_w[eg]);
        dg[h] = 1.0f / (1.0f + expf(-dec_hid[h])) - gs;
    }
}

// ------------------------------ main recurrence -----------------------------
__global__ __launch_bounds__(32) void fiuri_scan(
    const float*  __restrict__ x,        // (B,T,2)
    const float*  __restrict__ thr_hid,  // (64,)
    const float*  __restrict__ dec_out,  // (2,)
    const __bf16* __restrict__ bhidG,
    const __bf16* __restrict__ boutG,
    const float*  __restrict__ binfG,
    const float*  __restrict__ dgG,
    float*        __restrict__ out)      // (B,T,2)
{
    __shared__ __attribute__((aligned(16))) __bf16 lds[16][64]; // hid_O tile

    const int lane = threadIdx.x;        // wave32
    const int l16  = lane & 15;
    const int g    = lane >> 4;
    const int b0   = blockIdx.x * 16;

    // resident B matrices (pre-packed)
    v16bf Bh[8], Bo[2];
    v2f   Bi[4];
#pragma unroll
    for (int t = 0; t < 8; ++t) Bh[t] = *(const v16bf*)(bhidG + t * 512 + lane * 16);
#pragma unroll
    for (int t = 0; t < 2; ++t) Bo[t] = *(const v16bf*)(boutG + t * 512 + lane * 16);
#pragma unroll
    for (int t = 0; t < 4; ++t) Bi[t] = *(const v2f*)(binfG + t * 64 + lane * 2);

    float dg4[4], th4[4];
#pragma unroll
    for (int nt = 0; nt < 4; ++nt) {
        dg4[nt] = dgG[l16 + 16 * nt];        // D layout: h = l16 + 16*nt
        th4[nt] = thr_hid[l16 + 16 * nt];
    }
    float dout = (l16 < NOUT) ? 1.0f / (1.0f + expf(-dec_out[l16])) : 0.0f;

    // zero hid_O(t=-1)
    for (int i = lane; i < 16 * 64; i += 32) ((__bf16*)lds)[i] = (__bf16)0.0f;
    __syncthreads();

    v8f hidE[4];
    v8f outE;
#pragma unroll
    for (int nt = 0; nt < 4; ++nt)
#pragma unroll
        for (int r = 0; r < 8; ++r) hidE[nt][r] = 0.0f;
#pragma unroll
    for (int r = 0; r < 8; ++r) outE[r] = 0.0f;

    // one step of the recurrence
    auto step = [&](float x0, float x1, int t) {
        // in_O = (|x0|,|x1|,|x0|,|x1|): f32 A-tile for 16x16x4 is just
        // (|x0|,|x1|) on every lane (lanes 0-15 = K0,1; lanes 16-31 = K2,3).
        v2f ax;
        ax[0] = fabsf(x0);
        ax[1] = fabsf(x1);

        // old hid_O in A layout: lane-half g -> elems 0..15 K=16g..16g+15
        union { v16bf v; v8bf h[2]; } alo, ahi;
        alo.h[0] = *(const v8bf*)&lds[l16][ 0 + 16 * g];
        alo.h[1] = *(const v8bf*)&lds[l16][ 8 + 16 * g];
        ahi.h[0] = *(const v8bf*)&lds[l16][32 + 16 * g];
        ahi.h[1] = *(const v8bf*)&lds[l16][40 + 16 * g];

        // output layer (uses OLD hid_O): out_E = d_out*out_E + hid_O @ W_ho
        v8f co;
#pragma unroll
        for (int r = 0; r < 8; ++r) co[r] = dout * outE[r];
        co = __builtin_amdgcn_wmma_f32_16x16x32_bf16(false, alo.v, false, Bo[0],
                                                     (short)0, co, false, false);
        co = __builtin_amdgcn_wmma_f32_16x16x32_bf16(false, ahi.v, false, Bo[1],
                                                     (short)0, co, false, false);
        outE = co;

        // hidden update: E' = dg*E + in_O@(-W_ii+gjmat) + hid_O@(W_he-W_hi)
#pragma unroll
        for (int nt = 0; nt < 4; ++nt) {
            v8f c;
#pragma unroll
            for (int r = 0; r < 8; ++r) c[r] = dg4[nt] * hidE[nt][r];
            c = __builtin_amdgcn_wmma_f32_16x16x4_f32(false, ax, false, Bi[nt],
                                                      (short)0, c, false, false);
            c = __builtin_amdgcn_wmma_f32_16x16x32_bf16(false, alo.v, false, Bh[nt],
                                                        (short)0, c, false, false);
            c = __builtin_amdgcn_wmma_f32_16x16x32_bf16(false, ahi.v, false, Bh[4 + nt],
                                                        (short)0, c, false, false);
            hidE[nt] = c;
            // hid_O' = relu(E' - thr): D layout (M=r+8g, N=l16+16nt) -> LDS
#pragma unroll
            for (int r = 0; r < 8; ++r) {
                float o = c[r] - th4[nt];
                o = o > 0.0f ? o : 0.0f;
                lds[r + 8 * g][l16 + 16 * nt] = (__bf16)o;
            }
        }

        // a_t = tanh(out_E'); only N-cols 0,1 are real outputs
        if (l16 < NOUT) {
#pragma unroll
            for (int r = 0; r < 8; ++r) {
                int b = b0 + r + 8 * g;
                out[((size_t)b * TT + t) * 2 + l16] = tanhf(co[r]);
            }
        }
    };

    // ---- software-pipelined scan: chunks of 8 timesteps -------------------
    // Each lane streams its batch row (lanes 16-31 duplicate lanes 0-15;
    // harmless, keeps EXEC full).  float4 = 2 timesteps.
    const float4* xv = (const float4*)(x + (size_t)(b0 + l16) * TT * 2);

    float4 ca = xv[0], cb = xv[1], cc = xv[2], cd = xv[3];   // chunk 0
    for (int tc = 0; tc < TT / 8; ++tc) {
        const int tn = (tc + 1 < TT / 8) ? tc + 1 : tc;      // prefetch next
        float4 na = xv[4 * tn + 0];
        float4 nb = xv[4 * tn + 1];
        float4 nc = xv[4 * tn + 2];
        float4 nd = xv[4 * tn + 3];

        const int tbase = tc * 8;
        step(ca.x, ca.y, tbase + 0);
        step(ca.z, ca.w, tbase + 1);
        step(cb.x, cb.y, tbase + 2);
        step(cb.z, cb.w, tbase + 3);
        step(cc.x, cc.y, tbase + 4);
        step(cc.z, cc.w, tbase + 5);
        step(cd.x, cd.y, tbase + 6);
        step(cd.z, cd.w, tbase + 7);

        ca = na; cb = nb; cc = nc; cd = nd;
    }
}

// ------------------------------- launcher -----------------------------------
extern "C" void kernel_launch(void* const* d_in, const int* in_sizes, int n_in,
                              void* d_out, int out_size, void* d_ws, size_t ws_size,
                              hipStream_t stream) {
    (void)in_sizes; (void)n_in; (void)out_size; (void)ws_size;
    const float* x       = (const float*)d_in[0];
    const float* w_ii    = (const float*)d_in[1];
    const float* m_ii    = (const float*)d_in[2];
    const float* w_he    = (const float*)d_in[3];
    const float* m_he    = (const float*)d_in[4];
    const float* w_hi    = (const float*)d_in[5];
    const float* m_hi    = (const float*)d_in[6];
    const float* w_ho    = (const float*)d_in[7];
    const float* m_ho    = (const float*)d_in[8];
    const float* gj_w    = (const float*)d_in[9];
    const float* thr_hid = (const float*)d_in[10];
    const float* dec_hid = (const float*)d_in[11];
    const float* dec_out = (const float*)d_in[13];
    const int*   gj_src  = (const int*)d_in[14];
    const int*   gj_dst  = (const int*)d_in[15];

    __bf16* bhid = (__bf16*)d_ws;              // 8*512 bf16
    __bf16* bout = bhid + 8 * 512;             // 2*512 bf16
    float*  binf = (float*)(bout + 2 * 512);   // 4*64 f32 (8B-aligned)
    float*  dg   = binf + 4 * 64;              // 64 f32

    fiuri_pack<<<dim3(1), dim3(256), 0, stream>>>(
        w_ii, m_ii, w_he, m_he, w_hi, m_hi, w_ho, m_ho,
        gj_w, dec_hid, gj_src, gj_dst, bhid, bout, binf, dg);

    fiuri_scan<<<dim3(BB / 16), dim3(32), 0, stream>>>(
        x, thr_hid, dec_out, bhid, bout, binf, dg, (float*)d_out);
}
